// CausalSelfAttention_63617055588589
// MI455X (gfx1250) — compile-verified, matrix operand order checked by reading, not run on hardware
//
#include <hip/hip_runtime.h>
#include <hip/hip_bf16.h>
#include <math.h>

typedef _Float16 f16;
typedef __attribute__((ext_vector_type(8)))  _Float16 v8h;
typedef __attribute__((ext_vector_type(16))) _Float16 v16h;
typedef __attribute__((ext_vector_type(8)))  float    v8f;
typedef __attribute__((ext_vector_type(4)))  unsigned int v4u;
typedef __attribute__((ext_vector_type(8)))  unsigned int v8u;

#define NUM_HEADS 16
#define NUM_KV    4
#define HEAD_DIM  64
#define DIM       1024
#define KV_DIM    256
#define BATCH     2

static __device__ __forceinline__ v16h cat16(v8h lo, v8h hi) {
  return __builtin_shufflevector(lo, hi, 0,1,2,3,4,5,6,7,8,9,10,11,12,13,14,15);
}
static __device__ __forceinline__ v8f wmma_f16(v16h a, v16h b, v8f c) {
  return __builtin_amdgcn_wmma_f32_16x16x32_f16(false, a, false, b, (short)0, c, false, false);
}
// async global->LDS 16B copy (per lane), tracked by ASYNCcnt
static __device__ __forceinline__ void async_copy_b128(const f16* gptr, f16* lptr) {
  unsigned lds = (unsigned)(uintptr_t)lptr;                  // low 32 bits = LDS byte addr
  unsigned long long g = (unsigned long long)(uintptr_t)gptr;
  asm volatile("global_load_async_to_lds_b128 %0, %1, off" :: "v"(lds), "v"(g) : "memory");
}
static __device__ __forceinline__ void wait_async0() {
  asm volatile("s_wait_asynccnt 0" ::: "memory");
}

// Tensor Data Mover: 2D tile (f16) global -> LDS, one instruction per panel.
// Descriptor per CDNA5 ISA 8.3/8.4: group0 = {count/flags, lds_addr, gaddr lo,
// gaddr hi | type=2}; group1 = {data_size, tensor dims, tile dims, row stride}.
static __device__ __forceinline__ void tdm_load_2d_f16(const void* gaddr, unsigned lds_addr,
                                                       unsigned tile_w, unsigned tile_h,
                                                       unsigned long long row_stride_elems) {
  unsigned long long ga = (unsigned long long)(uintptr_t)gaddr;
  v4u g0;
  g0[0] = 1u;                                            // count=1, user D#
  g0[1] = lds_addr;                                      // LDS byte address
  g0[2] = (unsigned)ga;                                  // global addr [31:0]
  g0[3] = (unsigned)((ga >> 32) & 0x1FFFFFFu) | (2u << 30);  // addr[56:32] | type=2
  v8u g1;
  g1[0] = 0x00010000u;                                   // data_size = 2 bytes
  g1[1] = (tile_w & 0xFFFFu) << 16;                      // tensor_dim0[15:0]
  g1[2] = (tile_w >> 16) | ((tile_h & 0xFFFFu) << 16);   // td0[31:16] | td1[15:0]
  g1[3] = (tile_h >> 16) | (tile_w << 16);               // td1[31:16] | tile_dim0
  g1[4] = tile_h;                                        // tile_dim1 (tile_dim2=0)
  g1[5] = (unsigned)row_stride_elems;                    // dim0_stride[31:0]
  g1[6] = (unsigned)((row_stride_elems >> 32) & 0xFFFFu);// dim0_stride[47:32]
  g1[7] = 0u;
  asm volatile("tensor_load_to_lds %0, %1" :: "s"(g0), "s"(g1) : "memory");
}

// ---------------------------------------------------------------------------
// 1) int4 fake-quant of a weight matrix row (K=1024), f32 -> f16
// ---------------------------------------------------------------------------
__global__ __launch_bounds__(256) void quant_i4_kernel(const float* __restrict__ w,
                                                       f16* __restrict__ out, int rows) {
  int lane = threadIdx.x & 31, wave = threadIdx.x >> 5;
  int row = blockIdx.x * 8 + wave;
  if (row >= rows) return;
  const float* wr = w + (size_t)row * DIM;
  float mx = 0.f;
  #pragma unroll 4
  for (int j = 0; j < 32; ++j) mx = fmaxf(mx, fabsf(wr[lane + 32 * j]));
  #pragma unroll
  for (int off = 1; off < 32; off <<= 1) mx = fmaxf(mx, __shfl_xor(mx, off, 32));
  float scale = fmaxf(mx, 1e-8f) / 7.0f;
  float inv = 1.0f / scale;
  f16* orow = out + (size_t)row * DIM;
  #pragma unroll 4
  for (int j = 0; j < 32; ++j) {
    float v = wr[lane + 32 * j];
    float q = rintf(v * inv);              // round-half-even, matches jnp.round
    q = fminf(fmaxf(q, -7.f), 7.f);
    orow[lane + 32 * j] = (f16)(q * scale);
  }
}

// ---------------------------------------------------------------------------
// 2) f32 -> f16 convert
// ---------------------------------------------------------------------------
__global__ __launch_bounds__(256) void cvt_f32_f16_kernel(const float* __restrict__ in,
                                                          f16* __restrict__ out, int n) {
  int i = blockIdx.x * 256 + threadIdx.x;
  if (i < n) out[i] = (f16)in[i];
}

// ---------------------------------------------------------------------------
// 3) Tiled WMMA GEMM:  C[M,N](f32) = A[M,K](f16, row-major) * B[N,K](f16)^T
//    128x128 block, K-panel 32, 8 waves x (32x64). LDS double-buffered:
//    A panel via GLOBAL_LOAD_ASYNC_TO_LDS_B128 (ASYNCcnt, per-lane),
//    B panel via TENSOR_LOAD_TO_LDS (TDM, TENSORcnt, one instr per panel).
// ---------------------------------------------------------------------------
#define BM 128
#define BN 128
#define BK 32
__global__ __launch_bounds__(256) void gemm_wmma_kernel(const f16* __restrict__ A,
                                                        const f16* __restrict__ B,
                                                        float* __restrict__ C,
                                                        int M, int N, int K) {
  __shared__ __align__(16) f16 As[2][BM * BK];
  __shared__ __align__(16) f16 Bs[2][BN * BK];
  const int tid = threadIdx.x;
  const int lane = tid & 31;
  const int wave = tid >> 5;
  const int m0 = blockIdx.y * BM, n0 = blockIdx.x * BN;
  const int wm = wave >> 1;           // 0..3 -> 32-row strip
  const int wn = wave & 1;            // 0..1 -> 64-col strip
  const int cl = lane & 15;           // fragment row / column within tile
  const int kb = (lane >> 4) * 8;     // K sub-offset per half-wave

  v8f acc[2][4] = {};

  // stage one K-panel pair into buffer `buf`
  auto stage = [&](int buf, int k0) {
    // A panel: per-lane async copies (ASYNCcnt)
    #pragma unroll
    for (int c = tid; c < (BM * BK / 8); c += 256) {
      int r = c >> 2, cc = (c & 3) * 8;
      async_copy_b128(&A[(size_t)(m0 + r) * K + k0 + cc], &As[buf][r * BK + cc]);
    }
    // B panel: one TDM descriptor, issued by wave 0 (TENSORcnt)
    if (wave == 0) {
      tdm_load_2d_f16(&B[(size_t)n0 * K + k0],
                      (unsigned)(uintptr_t)&Bs[buf][0],
                      BK, BN, (unsigned long long)K);
    }
  };
  auto stage_wait = [&]() {
    wait_async0();                                   // A panel (all waves)
    if (wave == 0) __builtin_amdgcn_s_wait_tensorcnt(0);  // B panel (issuer)
  };

  const int nk = K / BK;
  stage(0, 0);
  stage_wait();
  __syncthreads();

  for (int kt = 0; kt < nk; ++kt) {
    const int buf = kt & 1;
    if (kt + 1 < nk) stage(buf ^ 1, (kt + 1) * BK);   // overlap with compute
    if (kt + 2 < nk) {                                // pull tile t+2 toward GL2
      __builtin_prefetch(&A[(size_t)(m0 + (tid >> 1)) * K + (kt + 2) * BK], 0, 1);
      __builtin_prefetch(&B[(size_t)(n0 + (tid >> 1)) * K + (kt + 2) * BK], 0, 1);
    }

    v16h af[2], bf[4];
    #pragma unroll
    for (int mi = 0; mi < 2; ++mi) {
      const f16* p = &As[buf][(wm * 32 + mi * 16 + cl) * BK + kb];
      af[mi] = cat16(*(const v8h*)p, *(const v8h*)(p + 16));
    }
    #pragma unroll
    for (int ni = 0; ni < 4; ++ni) {
      const f16* p = &Bs[buf][(wn * 64 + ni * 16 + cl) * BK + kb];
      bf[ni] = cat16(*(const v8h*)p, *(const v8h*)(p + 16));
    }
    #pragma unroll
    for (int mi = 0; mi < 2; ++mi)
      #pragma unroll
      for (int ni = 0; ni < 4; ++ni)
        acc[mi][ni] = wmma_f16(af[mi], bf[ni], acc[mi][ni]);

    stage_wait();      // next panel fully in LDS (this wave's copies)
    __syncthreads();   // ... and everyone else's; reads of `buf` also done
  }

  const int rowadd = (lane >> 4) * 8;  // C layout: lanes 16-31 hold rows +8
  #pragma unroll
  for (int mi = 0; mi < 2; ++mi)
    #pragma unroll
    for (int ni = 0; ni < 4; ++ni)
      #pragma unroll
      for (int e = 0; e < 8; ++e) {
        int grow = m0 + wm * 32 + mi * 16 + e + rowadd;
        int gcol = n0 + wn * 64 + ni * 16 + cl;
        C[(size_t)grow * N + gcol] = acc[mi][ni][e];
      }
}

// ---------------------------------------------------------------------------
// 4) per-(b,s,head) RMS-norm + RoPE (+gain for Q); (b,s,h,d)f32 -> (b,h,s,d)f16
// ---------------------------------------------------------------------------
__global__ __launch_bounds__(256) void norm_rope_kernel(const float* __restrict__ in,
                                                        f16* __restrict__ out,
                                                        const float* __restrict__ gain,
                                                        int nh, int S) {
  int lane = threadIdx.x & 31, wave = threadIdx.x >> 5;
  int row = blockIdx.x * 8 + wave;
  int total = BATCH * S * nh;
  if (row >= total) return;
  int b = row / (S * nh);
  int rem = row - b * S * nh;
  int s = rem / nh;
  int h = rem - s * nh;
  const float* ip = in + ((size_t)(b * S + s) * nh + h) * HEAD_DIM;
  float a = ip[lane], c2 = ip[lane + 32];
  float ss = a * a + c2 * c2;
  #pragma unroll
  for (int off = 1; off < 32; off <<= 1) ss += __shfl_xor(ss, off, 32);
  float rn = rsqrtf(ss * (1.0f / 64.0f) + 1.1920929e-07f);
  a *= rn; c2 *= rn;
  // inv_freq = 10000^(-2*lane/64)
  float ang = (float)s * __expf(-((float)(2 * lane) * (1.0f / 64.0f)) * 9.210340371976184f);
  float sn, cs;
  __sincosf(ang, &sn, &cs);
  float o1 = a * cs + c2 * sn;
  float o2 = -a * sn + c2 * cs;
  if (gain) { float g = gain[h]; o1 *= g; o2 *= g; }
  f16* op = out + ((size_t)(b * nh + h) * S + s) * HEAD_DIM;
  op[lane] = (f16)o1;
  op[lane + 32] = (f16)o2;
}

// ---------------------------------------------------------------------------
// 5) V: (b,s,h,d) f32 -> (b,h,s,d) f16
// ---------------------------------------------------------------------------
__global__ __launch_bounds__(256) void v_transpose_kernel(const float* __restrict__ in,
                                                          f16* __restrict__ out, int S) {
  int lane = threadIdx.x & 31, wave = threadIdx.x >> 5;
  int row = blockIdx.x * 8 + wave;
  int total = BATCH * S * NUM_KV;
  if (row >= total) return;
  int b = row / (S * NUM_KV);
  int rem = row - b * S * NUM_KV;
  int s = rem / NUM_KV;
  int h = rem - s * NUM_KV;
  const float* ip = in + ((size_t)(b * S + s) * NUM_KV + h) * HEAD_DIM;
  f16* op = out + ((size_t)(b * NUM_KV + h) * S + s) * HEAD_DIM;
  op[lane] = (f16)ip[lane];
  op[lane + 32] = (f16)ip[lane + 32];
}

// ---------------------------------------------------------------------------
// 6) Flash attention: one wave per (b, h, 16-query tile). Online softmax with
//    cross-lane reductions in the WMMA C layout; P relayout via LDS.
// ---------------------------------------------------------------------------
__global__ __launch_bounds__(256) void attn_wmma_kernel(const f16* __restrict__ qh,
                                                        const f16* __restrict__ kh,
                                                        const f16* __restrict__ vh,
                                                        f16* __restrict__ y, int S) {
  __shared__ __align__(16) f16 Pt[8][16 * 32];
  const int lane = threadIdx.x & 31, wave = threadIdx.x >> 5;
  const int qblocks = S >> 4;
  int task = blockIdx.x * 8 + wave;
  if (task >= BATCH * NUM_HEADS * qblocks) return;
  const int bh = task / qblocks;
  const int qb = task - bh * qblocks;
  const int b = bh / NUM_HEADS;
  const int h = bh - b * NUM_HEADS;
  const int kvh = h >> 2;  // rep = 4

  const f16* qptr = qh + (((size_t)(b * NUM_HEADS + h) * S) + qb * 16) * HEAD_DIM;
  const f16* kbas = kh + ((size_t)(b * NUM_KV + kvh) * S) * HEAD_DIM;
  const f16* vbas = vh + ((size_t)(b * NUM_KV + kvh) * S) * HEAD_DIM;

  const int cl = lane & 15;
  const int kboff = (lane >> 4) * 8;
  const int rowadd = (lane >> 4) * 8;

  // Q fragments (16x64 split into two 16x32 A-frags), pre-scaled by 1/sqrt(64)
  v16h qf[2];
  #pragma unroll
  for (int kc = 0; kc < 2; ++kc) {
    const f16* p = qptr + (size_t)cl * HEAD_DIM + kc * 32 + kboff;
    v16h q = cat16(*(const v8h*)p, *(const v8h*)(p + 16));
    #pragma unroll
    for (int i = 0; i < 16; ++i) q[i] = q[i] * (_Float16)0.125f;
    qf[kc] = q;
  }

  v8f acc[4] = {};
  float mrow[8], lrow[8];
  #pragma unroll
  for (int e = 0; e < 8; ++e) { mrow[e] = -1e30f; lrow[e] = 0.f; }

  const int nkb = (qb * 16 + 47) >> 5;  // causal: key blocks of 32
  for (int kb2 = 0; kb2 < nkb; ++kb2) {
    const int kstart = kb2 * 32;

    // scores S = Q K^T : two 16x16 tiles
    v8f sc[2];
    #pragma unroll
    for (int t = 0; t < 2; ++t) {
      const f16* kp = kbas + (size_t)(kstart + t * 16 + cl) * HEAD_DIM + kboff;
      v16h kf0 = cat16(*(const v8h*)kp,        *(const v8h*)(kp + 16));
      v16h kf1 = cat16(*(const v8h*)(kp + 32), *(const v8h*)(kp + 48));
      v8f s = {};
      s = wmma_f16(qf[0], kf0, s);
      s = wmma_f16(qf[1], kf1, s);
      #pragma unroll
      for (int e = 0; e < 8; ++e) {
        int rq = qb * 16 + e + rowadd;
        int ck = kstart + t * 16 + cl;
        if (ck > rq) s[e] = -1e30f;
      }
      sc[t] = s;
    }

    // online softmax per row (rows live per-VGPR-slot within each half-wave)
    float corr[8];
    #pragma unroll
    for (int e = 0; e < 8; ++e) {
      float mx = fmaxf(sc[0][e], sc[1][e]);
      #pragma unroll
      for (int off = 1; off < 16; off <<= 1) mx = fmaxf(mx, __shfl_xor(mx, off, 32));
      float nm = fmaxf(mrow[e], mx);
      corr[e] = __expf(mrow[e] - nm);
      mrow[e] = nm;
      sc[0][e] = __expf(sc[0][e] - nm);
      sc[1][e] = __expf(sc[1][e] - nm);
      float rs = sc[0][e] + sc[1][e];
      #pragma unroll
      for (int off = 1; off < 16; off <<= 1) rs += __shfl_xor(rs, off, 32);
      lrow[e] = lrow[e] * corr[e] + rs;
      #pragma unroll
      for (int nt = 0; nt < 4; ++nt) acc[nt][e] *= corr[e];
    }

    // relayout P (C layout -> A fragment) via this wave's LDS tile
    f16* pt = Pt[wave];
    #pragma unroll
    for (int t = 0; t < 2; ++t)
      #pragma unroll
      for (int e = 0; e < 8; ++e)
        pt[(e + rowadd) * 32 + t * 16 + cl] = (f16)sc[t][e];
    asm volatile("s_wait_dscnt 0" ::: "memory");
    const f16* pp = pt + cl * 32 + kboff;
    v16h pf = cat16(*(const v8h*)pp, *(const v8h*)(pp + 16));

    // acc += P (16x32) * V (32x64)
    #pragma unroll
    for (int nt = 0; nt < 4; ++nt) {
      const f16* vp = vbas + (size_t)(kstart + kboff) * HEAD_DIM + nt * 16 + cl;
      v8h lo = {}, hi = {};
      #pragma unroll
      for (int j = 0; j < 8; ++j) {
        lo[j] = vp[(size_t)j * HEAD_DIM];
        hi[j] = vp[(size_t)(16 + j) * HEAD_DIM];
      }
      acc[nt] = wmma_f16(pf, cat16(lo, hi), acc[nt]);
    }
  }

  // epilogue: normalize and store to (b, s, h*64) f16
  #pragma unroll
  for (int e = 0; e < 8; ++e) {
    float inv = 1.0f / lrow[e];
    int srow = qb * 16 + e + rowadd;
    #pragma unroll
    for (int nt = 0; nt < 4; ++nt) {
      y[((size_t)(b * S + srow) * DIM) + h * HEAD_DIM + nt * 16 + cl] =
          (f16)(acc[nt][e] * inv);
    }
  }
}

// ---------------------------------------------------------------------------
extern "C" void kernel_launch(void* const* d_in, const int* in_sizes, int n_in,
                              void* d_out, int out_size, void* d_ws, size_t ws_size,
                              hipStream_t stream) {
  const float* x    = (const float*)d_in[0];
  const float* w_q  = (const float*)d_in[1];
  const float* w_k  = (const float*)d_in[2];
  const float* w_v  = (const float*)d_in[3];
  const float* w_p  = (const float*)d_in[4];
  const float* gain = (const float*)d_in[5];
  float* out = (float*)d_out;

  const int BS = in_sizes[0] / DIM;   // batch * seqlen = 4096
  const int S  = BS / BATCH;          // 2048

  // carve workspace (all offsets 256B aligned)
  char* p = (char*)d_ws;
  size_t off = 0;
  auto alloc = [&](size_t bytes) { void* r = p + off; off = (off + bytes + 255) & ~(size_t)255; return r; };
  f16*  wqh = (f16*)alloc((size_t)DIM * DIM * 2);
  f16*  wkh = (f16*)alloc((size_t)KV_DIM * DIM * 2);
  f16*  wvh = (f16*)alloc((size_t)KV_DIM * DIM * 2);
  f16*  wph = (f16*)alloc((size_t)DIM * DIM * 2);
  f16*  xh  = (f16*)alloc((size_t)BS * DIM * 2);
  float* qf = (float*)alloc((size_t)BS * DIM * 4);
  float* kf = (float*)alloc((size_t)BS * KV_DIM * 4);
  float* vf = (float*)alloc((size_t)BS * KV_DIM * 4);
  f16*  q_h = (f16*)alloc((size_t)BS * DIM * 2);
  f16*  k_h = (f16*)alloc((size_t)BS * KV_DIM * 2);
  f16*  v_h = (f16*)alloc((size_t)BS * KV_DIM * 2);
  f16*  y_h = (f16*)alloc((size_t)BS * DIM * 2);
  (void)ws_size;

  // 1) fake-quant all weights to f16
  quant_i4_kernel<<<DIM / 8,    256, 0, stream>>>(w_q, wqh, DIM);
  quant_i4_kernel<<<KV_DIM / 8, 256, 0, stream>>>(w_k, wkh, KV_DIM);
  quant_i4_kernel<<<KV_DIM / 8, 256, 0, stream>>>(w_v, wvh, KV_DIM);
  quant_i4_kernel<<<DIM / 8,    256, 0, stream>>>(w_p, wph, DIM);

  // 2) x -> f16
  cvt_f32_f16_kernel<<<(BS * DIM + 255) / 256, 256, 0, stream>>>(x, xh, BS * DIM);

  // 3) QKV projections (WMMA GEMMs, async + TDM staged, double-buffered)
  gemm_wmma_kernel<<<dim3(DIM / BN,    BS / BM), 256, 0, stream>>>(xh, wqh, qf, BS, DIM,    DIM);
  gemm_wmma_kernel<<<dim3(KV_DIM / BN, BS / BM), 256, 0, stream>>>(xh, wkh, kf, BS, KV_DIM, DIM);
  gemm_wmma_kernel<<<dim3(KV_DIM / BN, BS / BM), 256, 0, stream>>>(xh, wvh, vf, BS, KV_DIM, DIM);

  // 4) RMS-norm + RoPE (+gain on Q), into (b,h,s,d) f16
  norm_rope_kernel<<<(BATCH * S * NUM_HEADS) / 8, 256, 0, stream>>>(qf, q_h, gain, NUM_HEADS, S);
  norm_rope_kernel<<<(BATCH * S * NUM_KV) / 8,    256, 0, stream>>>(kf, k_h, nullptr, NUM_KV, S);

  // 5) V -> (b,h,s,d) f16
  v_transpose_kernel<<<(BATCH * S * NUM_KV) / 8, 256, 0, stream>>>(vf, v_h, S);

  // 6) flash attention (one wave per 16-query tile)
  int tasks = BATCH * NUM_HEADS * (S / 16);
  attn_wmma_kernel<<<(tasks + 7) / 8, 256, 0, stream>>>(q_h, k_h, v_h, y_h, S);

  // 7) output projection -> f32 d_out
  gemm_wmma_kernel<<<dim3(DIM / BN, BS / BM), 256, 0, stream>>>(y_h, wph, out, BS, DIM, DIM);
}
